// SutraV054_34084860461122
// MI455X (gfx1250) — matrix operand order repaired and checked by reading, not compiled, not run on hardware
//
#include <hip/hip_runtime.h>
#include <hip/hip_bf16.h>
#include <math.h>

// ---------------- model constants ----------------
constexpr int cDIM   = 512;
constexpr int cFF    = 1024;
constexpr int cS     = 5;
constexpr int cT     = 512;
constexpr int cB     = 8;
constexpr int cM     = cB * cT;      // 4096 rows
constexpr int cNSLOT = 8;
constexpr int cVOCAB = 32000;
constexpr int cWIN   = 16;
constexpr int cKRET  = 4;
constexpr int cSTEPS = 6;

constexpr float cRSQRT_DIM = 0.04419417382415922f;  // 1/sqrt(512)

// activation codes for the GEMM epilogue
constexpr int ACT_NONE     = 0;
constexpr int ACT_SILU     = 1;
constexpr int ACT_SOFTPLUS01 = 2;   // softplus(x)+0.1
constexpr int ACT_SOFTPLUS = 3;

typedef _Float16 v16h __attribute__((ext_vector_type(16)));
typedef float    v8f  __attribute__((ext_vector_type(8)));

__device__ __forceinline__ float sutra_act(float x, int act) {
  if (act == ACT_SILU)       return x / (1.f + expf(-x));
  if (act == ACT_SOFTPLUS01) { float sp = (x > 20.f) ? x : log1pf(expf(x)); return sp + 0.1f; }
  if (act == ACT_SOFTPLUS)   return (x > 20.f) ? x : log1pf(expf(x));
  return x;
}

// =====================================================================
// Generic WMMA GEMM:  C[M,N] = act(A[M,K] @ W + bias) * scale
//   transB==0 : W is [K,N] row-major
//   transB==1 : W is [N,K] row-major  (used for logits = final @ emb.T)
// 256 threads = 8 wave32s; workgroup tile 64x64; K staged 32 at a time
// through LDS with f32 -> f16 conversion; f32 accumulation in WMMA.
// =====================================================================
__global__ __launch_bounds__(256)
void sutra_gemm_wmma(const float* __restrict__ A, const float* __restrict__ W,
                     const float* __restrict__ bias, float* __restrict__ C,
                     int M_, int N_, int K_, int transB, int act, float scale) {
  __shared__ _Float16 As[64][34];   // 64 rows x 32 k (pad 2)
  __shared__ _Float16 Bs[32][72];   // 32 k x 64 n (pad 8)

  const int tid  = threadIdx.x;
  const int lane = tid & 31;
  const int wave = tid >> 5;
  const int row0 = blockIdx.y * 64;
  const int col0 = blockIdx.x * 64;
  const int wm   = wave & 3;        // 4 waves along M (16 rows each)
  const int wn   = wave >> 2;       // 2 waves along N (32 cols each)

  v8f acc0 = {0.f,0.f,0.f,0.f,0.f,0.f,0.f,0.f};
  v8f acc1 = {0.f,0.f,0.f,0.f,0.f,0.f,0.f,0.f};

  for (int k0 = 0; k0 < K_; k0 += 32) {
    // ---- stage A tile 64x32 (4 threads per row, 8 elems each) ----
    {
      int r  = tid >> 2;
      int c0 = (tid & 3) << 3;
      int gr = row0 + r;
      if (gr < M_ && k0 + 32 < K_)
        __builtin_prefetch(&A[(size_t)gr * K_ + k0 + 32 + c0], 0, 3);
#pragma unroll
      for (int j = 0; j < 8; ++j) {
        int gk = k0 + c0 + j;
        float vA = (gr < M_ && gk < K_) ? A[(size_t)gr * K_ + gk] : 0.f;
        As[r][c0 + j] = (_Float16)vA;
      }
    }
    // ---- stage B tile 32x64 ----
    if (!transB) {
      int r   = tid >> 3;            // k within chunk
      int c0n = (tid & 7) << 3;
      int gk  = k0 + r;
#pragma unroll
      for (int j = 0; j < 8; ++j) {
        int gn = col0 + c0n + j;
        float vB = (gk < K_ && gn < N_) ? W[(size_t)gk * N_ + gn] : 0.f;
        Bs[r][c0n + j] = (_Float16)vB;
      }
    } else {
      int nn = tid & 63;
      int kb = (tid >> 6) << 3;
      int gn = col0 + nn;
#pragma unroll
      for (int j = 0; j < 8; ++j) {
        int gk = k0 + kb + j;
        float vB = (gn < N_ && gk < K_) ? W[(size_t)gn * K_ + gk] : 0.f;
        Bs[kb + j][nn] = (_Float16)vB;
      }
    }
    __syncthreads();

    // ---- A fragment (contiguous k = khi + hh, same order as B) ----
    v16h afrag;
    {
      int mrow = (wm << 4) + (lane & 15);
      int khi  = (lane >> 4) << 4;          // 0 or 16
#pragma unroll
      for (int hh = 0; hh < 8; ++hh)  afrag[hh] = As[mrow][khi + hh];
#pragma unroll
      for (int hh = 8; hh < 16; ++hh) afrag[hh] = As[mrow][khi + hh]; // K = khi + hh
    }
    // ---- B fragments (lane = N, lane-group = K half) ----
    v16h bfrag0, bfrag1;
    {
      int kb   = (lane >> 4) << 4;          // 0 or 16
      int ncol = (wn << 5) + (lane & 15);
#pragma unroll
      for (int hh = 0; hh < 16; ++hh) {
        bfrag0[hh] = Bs[kb + hh][ncol];
        bfrag1[hh] = Bs[kb + hh][ncol + 16];
      }
    }
    acc0 = __builtin_amdgcn_wmma_f32_16x16x32_f16(false, afrag, false, bfrag0,
                                                  (short)0, acc0, false, false);
    acc1 = __builtin_amdgcn_wmma_f32_16x16x32_f16(false, afrag, false, bfrag1,
                                                  (short)0, acc1, false, false);
    __syncthreads();
  }

  // ---- epilogue: D layout lane=N, VGPR r -> M = r + 8*(lane>=16) ----
  int mbase = row0 + (wm << 4) + ((lane >> 4) << 3);
  int nb0   = col0 + (wn << 5) + (lane & 15);
#pragma unroll
  for (int r = 0; r < 8; ++r) {
    int gm = mbase + r;
    if (gm >= M_) continue;
    if (nb0 < N_) {
      float vo = acc0[r] + (bias ? bias[nb0] : 0.f);
      C[(size_t)gm * N_ + nb0] = sutra_act(vo, act) * scale;
    }
    if (nb0 + 16 < N_) {
      float vo = acc1[r] + (bias ? bias[nb0 + 16] : 0.f);
      C[(size_t)gm * N_ + nb0 + 16] = sutra_act(vo, act) * scale;
    }
  }
}

// =====================================================================
// Elementwise / reduction helper kernels
// =====================================================================

__global__ void sutra_embed_kernel(const int* __restrict__ x, const float* __restrict__ emb,
                                   const float* __restrict__ pos, float* __restrict__ h) {
  int i = blockIdx.x * blockDim.x + threadIdx.x;
  if (i >= cM * cDIM) return;
  int m = i / cDIM, d = i % cDIM, t = m % cT;
  h[i] = emb[(size_t)x[m] * cDIM + d] + pos[(size_t)t * cDIM + d];
}

__global__ void sutra_init_pi_kernel(float* __restrict__ pi) {
  int i = blockIdx.x * blockDim.x + threadIdx.x;
  if (i >= cM * cS) return;
  pi[i] = ((i % cS) == 2) ? 1.f : 0.f;
}

__global__ void sutra_init_mem_kernel(const float* __restrict__ sp_mem, float* __restrict__ mem) {
  int i = blockIdx.x * blockDim.x + threadIdx.x;
  if (i >= cB * cNSLOT * cDIM) return;
  mem[i] = sp_mem[i % (cNSLOT * cDIM)];
}

// wave-per-row layernorm (wave32); safe in-place
__global__ void sutra_layernorm_kernel(const float* __restrict__ x, float* __restrict__ y,
                                       const float* __restrict__ g, const float* __restrict__ b) {
  int wave = threadIdx.x >> 5, lane = threadIdx.x & 31;
  int m = blockIdx.x * (blockDim.x >> 5) + wave;
  if (m >= cM) return;
  const float* row = x + (size_t)m * cDIM;
  float xv[cDIM / 32];
  float s = 0.f;
#pragma unroll
  for (int i = 0; i < cDIM / 32; ++i) { xv[i] = row[lane + i * 32]; s += xv[i]; }
  for (int off = 16; off > 0; off >>= 1) s += __shfl_xor(s, off, 32);
  float mean = s / (float)cDIM;
  float v = 0.f;
#pragma unroll
  for (int i = 0; i < cDIM / 32; ++i) { float d = xv[i] - mean; v += d * d; }
  for (int off = 16; off > 0; off >>= 1) v += __shfl_xor(v, off, 32);
  float r = rsqrtf(v / (float)cDIM + 1e-5f);
#pragma unroll
  for (int i = 0; i < cDIM / 32; ++i) {
    int d = lane + i * 32;
    y[(size_t)m * cDIM + d] = (xv[i] - mean) * r * g[d] + b[d];
  }
}

// base = hid @ tr_w2 + tr_b2   (N = 25, too small for WMMA tiles)
__global__ void sutra_tr2_kernel(const float* __restrict__ hid, const float* __restrict__ w,
                                 const float* __restrict__ b2, float* __restrict__ out) {
  int i = blockIdx.x * blockDim.x + threadIdx.x;
  if (i >= cM * cS * cS) return;
  int m = i / (cS * cS), o = i % (cS * cS);
  const float* hrow = hid + (size_t)m * cFF;
  float acc = b2[o];
  for (int k = 0; k < cFF; ++k) acc += hrow[k] * w[(size_t)k * (cS * cS) + o];
  out[i] = acc;
}

// per-batch gate: mean(mu) -> silu(@gate_w1) -> @gate_w2 -> softmax -> mode bias
__global__ void sutra_gate_mode_kernel(const float* __restrict__ mu,
                                       const float* __restrict__ gw1, const float* __restrict__ gb1,
                                       const float* __restrict__ gw2, const float* __restrict__ gb2,
                                       const float* __restrict__ mode_bias, float* __restrict__ mode) {
  int b = blockIdx.x, tid = threadIdx.x;
  __shared__ float mean[cDIM];
  __shared__ float hid[32];
  __shared__ float mix[2];
  for (int d = tid; d < cDIM; d += blockDim.x) {
    float s = 0.f;
    const float* base = mu + (size_t)(b * cT) * cDIM + d;
    for (int t = 0; t < cT; ++t) s += base[(size_t)t * cDIM];
    mean[d] = s / (float)cT;
  }
  __syncthreads();
  if (tid < 32) {
    float s = gb1[tid];
    for (int d = 0; d < cDIM; ++d) s += mean[d] * gw1[(size_t)d * 32 + tid];
    hid[tid] = s / (1.f + expf(-s));
  }
  __syncthreads();
  if (tid == 0) {
    float l0 = gb2[0], l1 = gb2[1];
    for (int s2 = 0; s2 < 32; ++s2) { l0 += hid[s2] * gw2[s2 * 2]; l1 += hid[s2] * gw2[s2 * 2 + 1]; }
    float mx = fmaxf(l0, l1);
    float e0 = expf(l0 - mx), e1 = expf(l1 - mx);
    mix[0] = e0 / (e0 + e1); mix[1] = e1 / (e0 + e1);
  }
  __syncthreads();
  if (tid < cS * cS)
    mode[b * cS * cS + tid] = mix[0] * mode_bias[tid] + mix[1] * mode_bias[cS * cS + tid];
}

// K = softmax(SG ? base+mode : NEG);  pi_ev = pi @ K
__global__ void sutra_k_pi_ev_kernel(const float* __restrict__ base, const float* __restrict__ mode,
                                     const float* __restrict__ pi, float* __restrict__ pi_ev) {
  int m = blockIdx.x * blockDim.x + threadIdx.x;
  if (m >= cM) return;
  int b = m / cT;
  float Kmat[cS][cS];
  for (int i = 0; i < cS; ++i) {
    float vals[cS]; float mx = -1e30f;
    for (int j = 0; j < cS; ++j) {
      bool sg = (j == i) || (j == ((i + 1) % cS)) || (j == 2);
      float v = sg ? (base[(size_t)m * 25 + i * 5 + j] + mode[b * 25 + i * 5 + j]) : -1e9f;
      vals[j] = v; mx = fmaxf(mx, v);
    }
    float ss = 0.f;
    for (int j = 0; j < cS; ++j) { vals[j] = expf(vals[j] - mx); ss += vals[j]; }
    for (int j = 0; j < cS; ++j) Kmat[i][j] = vals[j] / ss;
  }
  for (int j = 0; j < cS; ++j) {
    float s = 0.f;
    for (int i = 0; i < cS; ++i) s += pi[(size_t)m * cS + i] * Kmat[i][j];
    pi_ev[(size_t)m * cS + j] = s;
  }
}

// so (+)= pi[:,s] * outs
__global__ void sutra_accum_so_kernel(const float* __restrict__ outs, const float* __restrict__ pi,
                                      float* __restrict__ so, int sidx, int init) {
  int i = blockIdx.x * blockDim.x + threadIdx.x;
  if (i >= cM * cDIM) return;
  int m = i / cDIM;
  float v = pi[(size_t)m * cS + sidx] * outs[i];
  so[i] = init ? v : so[i] + v;
}

// ev = mu_b @ ev_w + ev_b   (N = 5)
__global__ void sutra_ev_kernel(const float* __restrict__ mu_b, const float* __restrict__ w,
                                const float* __restrict__ bb, float* __restrict__ ev) {
  int i = blockIdx.x * blockDim.x + threadIdx.x;
  if (i >= cM * cS) return;
  int m = i / cS, s2 = i % cS;
  const float* r = mu_b + (size_t)m * cDIM;
  float acc = bb[s2];
  for (int k = 0; k < cDIM; ++k) acc += r[k] * w[(size_t)k * cS + s2];
  ev[i] = acc;
}

// pi = top2_project( normalize( pi_ev * softmax(ev/0.5) ) )
__global__ void sutra_pi_update_kernel(const float* __restrict__ pi_ev, const float* __restrict__ ev,
                                       float* __restrict__ pi) {
  int m = blockIdx.x * blockDim.x + threadIdx.x;
  if (m >= cM) return;
  float e[cS]; float mx = -1e30f;
  for (int s2 = 0; s2 < cS; ++s2) { e[s2] = ev[(size_t)m * cS + s2] * 2.f; mx = fmaxf(mx, e[s2]); }
  float ss = 0.f;
  for (int s2 = 0; s2 < cS; ++s2) { e[s2] = expf(e[s2] - mx); ss += e[s2]; }
  float pn[cS]; float tot = 0.f;
  for (int s2 = 0; s2 < cS; ++s2) { pn[s2] = pi_ev[(size_t)m * cS + s2] * e[s2] / ss; tot += pn[s2]; }
  tot = fmaxf(tot, 1e-8f);
  for (int s2 = 0; s2 < cS; ++s2) pn[s2] /= tot;
  int i1 = 0;
  for (int s2 = 1; s2 < cS; ++s2) if (pn[s2] > pn[i1]) i1 = s2;
  int i2 = (i1 == 0) ? 1 : 0;
  for (int s2 = 0; s2 < cS; ++s2) if (s2 != i1 && pn[s2] > pn[i2]) i2 = s2;
  float s12 = fmaxf(pn[i1] + pn[i2], 1e-8f);
  for (int s2 = 0; s2 < cS; ++s2)
    pi[(size_t)m * cS + s2] = (s2 == i1 || s2 == i2) ? pn[s2] / s12 : 0.f;
}

// banded (±16) attention, top-4, softmax, gather V; msg = out * pi[:,3]
__global__ void sutra_router_kernel(const float* __restrict__ q, const float* __restrict__ k,
                                    const float* __restrict__ v, const float* __restrict__ pi,
                                    float* __restrict__ msg) {
  int wave = threadIdx.x >> 5, lane = threadIdx.x & 31;
  int m = blockIdx.x * 8 + wave;
  if (m >= cM) return;
  int b = m / cT, t = m % cT;
  int s0 = t - cWIN; if (s0 < 0) s0 = 0;
  int s1 = t + cWIN; if (s1 > cT - 1) s1 = cT - 1;
  int nc = s1 - s0 + 1;                         // 17..33, always >= KRET
  float sc[2 * cWIN + 1];
  const float* qrow = q + (size_t)m * cDIM;
  for (int c = 0; c < nc; ++c) {
    const float* krow = k + (size_t)(b * cT + s0 + c) * cDIM;
    float p = 0.f;
    for (int d = lane; d < cDIM; d += 32) p += qrow[d] * krow[d];
    for (int off = 16; off > 0; off >>= 1) p += __shfl_xor(p, off, 32);
    sc[c] = p * cRSQRT_DIM;
  }
  int  ti[cKRET]; float tv[cKRET];
  bool used[2 * cWIN + 1];
  for (int c = 0; c < nc; ++c) used[c] = false;
  for (int r = 0; r < cKRET; ++r) {
    int best = -1; float bv = -1e30f;
    for (int c = 0; c < nc; ++c)
      if (!used[c] && sc[c] > bv) { bv = sc[c]; best = c; }
    used[best] = true; ti[r] = s0 + best; tv[r] = bv;
  }
  float mxv = tv[0];
  float at[cKRET]; float asum = 0.f;
  for (int r = 0; r < cKRET; ++r) { at[r] = expf(tv[r] - mxv); asum += at[r]; }
  for (int r = 0; r < cKRET; ++r) at[r] /= asum;
  float pi3 = pi[(size_t)m * cS + 3];
  for (int d = lane; d < cDIM; d += 32) {
    float acc = 0.f;
    for (int r = 0; r < cKRET; ++r) acc += at[r] * v[(size_t)(b * cT + ti[r]) * cDIM + d];
    msg[(size_t)m * cDIM + d] = acc * pi3;
  }
}

// softmax attention over 8 memory slots
__global__ void sutra_sp_read_kernel(const float* __restrict__ q, const float* __restrict__ mem,
                                     float* __restrict__ mem_ctx) {
  int wave = threadIdx.x >> 5, lane = threadIdx.x & 31;
  int m = blockIdx.x * 8 + wave;
  if (m >= cM) return;
  int b = m / cT;
  const float* qrow = q + (size_t)m * cDIM;
  float sc[cNSLOT];
  for (int n = 0; n < cNSLOT; ++n) {
    const float* mr = mem + (size_t)(b * cNSLOT + n) * cDIM;
    float p = 0.f;
    for (int d = lane; d < cDIM; d += 32) p += qrow[d] * mr[d];
    for (int off = 16; off > 0; off >>= 1) p += __shfl_xor(p, off, 32);
    sc[n] = p * cRSQRT_DIM;
  }
  float mx = sc[0];
  for (int n = 1; n < cNSLOT; ++n) mx = fmaxf(mx, sc[n]);
  float ss = 0.f;
  for (int n = 0; n < cNSLOT; ++n) { sc[n] = expf(sc[n] - mx); ss += sc[n]; }
  for (int n = 0; n < cNSLOT; ++n) sc[n] /= ss;
  for (int d = lane; d < cDIM; d += 32) {
    float acc = 0.f;
    for (int n = 0; n < cNSLOT; ++n)
      acc += sc[n] * mem[(size_t)(b * cNSLOT + n) * cDIM + d];
    mem_ctx[(size_t)m * cDIM + d] = acc;
  }
}

__global__ void sutra_add_memctx_kernel(const float* __restrict__ mem_ctx, const float* __restrict__ pi,
                                        float* __restrict__ msg) {
  int i = blockIdx.x * blockDim.x + threadIdx.x;
  if (i >= cM * cDIM) return;
  int m = i / cDIM;
  msg[i] += 0.1f * mem_ctx[i] * pi[(size_t)m * cS + 3];
}

// Bayesian-style mu/lam update
__global__ void sutra_mu_lam_update_kernel(const float* __restrict__ mu_w, const float* __restrict__ lam_obs,
                                           const float* __restrict__ val, const float* __restrict__ pi,
                                           float* __restrict__ mu, float* __restrict__ lam) {
  int i = blockIdx.x * blockDim.x + threadIdx.x;
  if (i >= cM * cDIM) return;
  int m = i / cDIM;
  float pw = pi[(size_t)m * cS + 4];
  float lo = lam_obs[i];
  float l  = lam[i];
  float ln = l + pw * lo;
  mu[i]  = (l * mu_w[i] + pw * lo * val[i]) / ln;
  lam[i] = ln;
}

__global__ void sutra_add_so_kernel(const float* __restrict__ so, float* __restrict__ mu) {
  int i = blockIdx.x * blockDim.x + threadIdx.x;
  if (i >= cM * cDIM) return;
  mu[i] += 0.1f * so[i];
}

// summ[b] = sum_t(mu*pw) / max(sum_t pw, 1e-6)
__global__ void sutra_sp_summ_kernel(const float* __restrict__ mu, const float* __restrict__ pi,
                                     float* __restrict__ summ) {
  int b = blockIdx.x, tid = threadIdx.x;
  __shared__ float red[256];
  __shared__ float denom;
  float ps = 0.f;
  for (int t = tid; t < cT; t += 256) ps += pi[(size_t)(b * cT + t) * cS + 4];
  red[tid] = ps; __syncthreads();
  for (int s2 = 128; s2 > 0; s2 >>= 1) { if (tid < s2) red[tid] += red[tid + s2]; __syncthreads(); }
  if (tid == 0) denom = fmaxf(red[0], 1e-6f);
  __syncthreads();
  for (int d = tid; d < cDIM; d += 256) {
    float s2 = 0.f;
    for (int t = 0; t < cT; ++t)
      s2 += mu[(size_t)(b * cT + t) * cDIM + d] * pi[(size_t)(b * cT + t) * cS + 4];
    summ[(size_t)b * cDIM + d] = s2 / denom;
  }
}

// mem_out = 0.9*mem_in + 0.1 * sigmoid([mem_in,summ]@gw + gb) * (summ@vw + vb)
__global__ void sutra_sp_write_kernel(const float* __restrict__ mem_in, const float* __restrict__ summ,
                                      const float* __restrict__ gw, const float* __restrict__ gb,
                                      const float* __restrict__ vw, const float* __restrict__ vb,
                                      float* __restrict__ mem_out) {
  int i = blockIdx.x * blockDim.x + threadIdx.x;
  if (i >= cB * cNSLOT * cDIM) return;
  int d = i % cDIM;
  int bn = i / cDIM;
  int b  = bn / cNSLOT;
  const float* mrow = mem_in + (size_t)bn * cDIM;
  const float* srow = summ + (size_t)b * cDIM;
  float g = gb[d];
  for (int k = 0; k < cDIM; ++k) g += mrow[k] * gw[(size_t)k * cDIM + d];
  for (int k = 0; k < cDIM; ++k) g += srow[k] * gw[(size_t)(cDIM + k) * cDIM + d];
  g = 1.f / (1.f + expf(-g));
  float nv = vb[d];
  for (int k = 0; k < cDIM; ++k) nv += srow[k] * vw[(size_t)k * cDIM + d];
  mem_out[i] = 0.9f * mrow[d] + 0.1f * g * nv;
}

// =====================================================================
// Host orchestration
// =====================================================================
static void sutra_gemm(const float* A, const float* W, const float* bias, float* C,
                       int M_, int N_, int K_, int transB, int act, float scale,
                       hipStream_t s) {
  dim3 grid((N_ + 63) / 64, (M_ + 63) / 64);
  sutra_gemm_wmma<<<grid, 256, 0, s>>>(A, W, bias, C, M_, N_, K_, transB, act, scale);
}

static void sutra_layernorm(const float* x, float* y, const float* g, const float* b, hipStream_t s) {
  sutra_layernorm_kernel<<<cM / 8, 256, 0, s>>>(x, y, g, b);
}

extern "C" void kernel_launch(void* const* d_in, const int* in_sizes, int n_in,
                              void* d_out, int out_size, void* d_ws, size_t ws_size,
                              hipStream_t stream) {
  (void)in_sizes; (void)n_in; (void)out_size; (void)ws_size;

  const int*   x          = (const int*)d_in[0];
  const float* p_emb      = (const float*)d_in[1];
  const float* p_pos      = (const float*)d_in[2];
  const float* p_init_mu_w  = (const float*)d_in[3];
  const float* p_init_mu_b  = (const float*)d_in[4];
  const float* p_init_lam_w = (const float*)d_in[5];
  const float* p_init_lam_b = (const float*)d_in[6];
  const float* p_tr_w1    = (const float*)d_in[7];
  const float* p_tr_b1    = (const float*)d_in[8];
  const float* p_tr_w2    = (const float*)d_in[9];
  const float* p_tr_b2    = (const float*)d_in[10];
  const float* p_gate_w1  = (const float*)d_in[11];
  const float* p_gate_b1  = (const float*)d_in[12];
  const float* p_gate_w2  = (const float*)d_in[13];
  const float* p_gate_b2  = (const float*)d_in[14];
  const float* p_mode_bias= (const float*)d_in[15];
  const float* p_bank_w1  = (const float*)d_in[16];
  const float* p_bank_b1  = (const float*)d_in[17];
  const float* p_bank_w2  = (const float*)d_in[18];
  const float* p_bank_b2  = (const float*)d_in[19];
  const float* p_ev_w     = (const float*)d_in[20];
  const float* p_ev_b     = (const float*)d_in[21];
  const float* p_rq_w     = (const float*)d_in[22];
  const float* p_rq_b     = (const float*)d_in[23];
  const float* p_rk_w     = (const float*)d_in[24];
  const float* p_rk_b     = (const float*)d_in[25];
  const float* p_rv_w     = (const float*)d_in[26];
  const float* p_rv_b     = (const float*)d_in[27];
  const float* p_wr_lam_w = (const float*)d_in[28];
  const float* p_wr_lam_b = (const float*)d_in[29];
  const float* p_wr_val_w = (const float*)d_in[30];
  const float* p_wr_val_b = (const float*)d_in[31];
  const float* p_sp_mem   = (const float*)d_in[32];
  const float* p_sp_read_w= (const float*)d_in[33];
  const float* p_sp_read_b= (const float*)d_in[34];
  const float* p_sp_gate_w= (const float*)d_in[35];
  const float* p_sp_gate_b= (const float*)d_in[36];
  const float* p_sp_val_w = (const float*)d_in[37];
  const float* p_sp_val_b = (const float*)d_in[38];
  const float* p_ln_g    = (const float*)d_in[39];
  const float* p_ln_b    = (const float*)d_in[40];
  const float* p_preb_g  = (const float*)d_in[41];
  const float* p_preb_b  = (const float*)d_in[42];
  const float* p_postb_g = (const float*)d_in[43];
  const float* p_postb_b = (const float*)d_in[44];
  const float* p_prer_g  = (const float*)d_in[45];
  const float* p_prer_b  = (const float*)d_in[46];
  const float* p_postr_g = (const float*)d_in[47];
  const float* p_postr_b = (const float*)d_in[48];
  const float* p_prew_g  = (const float*)d_in[49];
  const float* p_prew_b  = (const float*)d_in[50];
  const float* p_postw_g = (const float*)d_in[51];
  const float* p_postw_b = (const float*)d_in[52];

  // ---------------- workspace carve-out (floats) ----------------
  float* ws = (float*)d_ws;
  size_t off = 0;
  auto alloc = [&](size_t n) { float* p = ws + off; off += n; return p; };
  const size_t MD = (size_t)cM * cDIM;

  float* h        = alloc(MD);
  float* mu       = alloc(MD);
  float* lam      = alloc(MD);
  float* mu_b     = alloc(MD);
  float* mu_n     = alloc(MD);
  float* tmp1     = alloc((size_t)cM * cFF);
  float* outs_tmp = alloc(MD);
  float* so       = alloc(MD);
  float* qb       = alloc(MD);
  float* kb       = alloc(MD);
  float* vb       = alloc(MD);
  float* msg      = alloc(MD);
  float* mem_ctx  = alloc(MD);
  float* lam_obs  = alloc(MD);
  float* val      = alloc(MD);
  float* base     = alloc((size_t)cM * cS * cS);
  float* pi       = alloc((size_t)cM * cS);
  float* pi_ev    = alloc((size_t)cM * cS);
  float* ev       = alloc((size_t)cM * cS);
  float* mode     = alloc((size_t)cB * cS * cS);
  float* mem_a    = alloc((size_t)cB * cNSLOT * cDIM);
  float* mem_b2   = alloc((size_t)cB * cNSLOT * cDIM);
  float* summ     = alloc((size_t)cB * cDIM);

  const int MDgrid = (int)((MD + 255) / 256);

  // ---------------- init ----------------
  sutra_embed_kernel<<<MDgrid, 256, 0, stream>>>(x, p_emb, p_pos, h);
  sutra_gemm(h, p_init_mu_w,  p_init_mu_b,  mu,  cM, cDIM, cDIM, 0, ACT_NONE,      1.f, stream);
  sutra_gemm(h, p_init_lam_w, p_init_lam_b, lam, cM, cDIM, cDIM, 0, ACT_SOFTPLUS01, 1.f, stream);
  sutra_init_pi_kernel<<<(cM * cS + 255) / 256, 256, 0, stream>>>(pi);
  sutra_init_mem_kernel<<<(cB * cNSLOT * cDIM + 255) / 256, 256, 0, stream>>>(p_sp_mem, mem_a);

  float* mem_cur = mem_a;
  float* mem_nxt = mem_b2;

  // ---------------- iterative refinement ----------------
  for (int step = 0; step < cSTEPS; ++step) {
    // transition logits
    sutra_gemm(mu, p_tr_w1, p_tr_b1, tmp1, cM, 2 * cDIM, cDIM, 0, ACT_SILU, 1.f, stream);
    sutra_tr2_kernel<<<(cM * cS * cS + 255) / 256, 256, 0, stream>>>(tmp1, p_tr_w2, p_tr_b2, base);
    sutra_gate_mode_kernel<<<cB, 256, 0, stream>>>(mu, p_gate_w1, p_gate_b1, p_gate_w2, p_gate_b2,
                                                   p_mode_bias, mode);
    sutra_k_pi_ev_kernel<<<(cM + 255) / 256, 256, 0, stream>>>(base, mode, pi, pi_ev);

    // expert banks
    sutra_layernorm(mu, mu_b, p_preb_g, p_preb_b, stream);
    for (int s = 0; s < cS; ++s) {
      sutra_gemm(mu_b, p_bank_w1 + (size_t)s * cDIM * cFF, p_bank_b1 + (size_t)s * cFF,
                 tmp1, cM, cFF, cDIM, 0, ACT_SILU, 1.f, stream);
      sutra_gemm(tmp1, p_bank_w2 + (size_t)s * cFF * cDIM, p_bank_b2 + (size_t)s * cDIM,
                 outs_tmp, cM, cDIM, cFF, 0, ACT_NONE, 1.f, stream);
      sutra_accum_so_kernel<<<MDgrid, 256, 0, stream>>>(outs_tmp, pi, so, s, s == 0);
    }
    sutra_ev_kernel<<<(cM * cS + 255) / 256, 256, 0, stream>>>(mu_b, p_ev_w, p_ev_b, ev);
    sutra_layernorm(so, so, p_postb_g, p_postb_b, stream);
    sutra_pi_update_kernel<<<(cM + 255) / 256, 256, 0, stream>>>(pi_ev, ev, pi);

    // local routing
    sutra_layernorm(mu, mu_n, p_prer_g, p_prer_b, stream);
    sutra_gemm(mu_n, p_rq_w, p_rq_b, qb, cM, cDIM, cDIM, 0, ACT_NONE, 1.f, stream);
    sutra_gemm(mu_n, p_rk_w, p_rk_b, kb, cM, cDIM, cDIM, 0, ACT_NONE, 1.f, stream);
    sutra_gemm(mu_n, p_rv_w, p_rv_b, vb, cM, cDIM, cDIM, 0, ACT_NONE, 1.f, stream);
    sutra_router_kernel<<<cM / 8, 256, 0, stream>>>(qb, kb, vb, pi, msg);
    sutra_layernorm(msg, msg, p_postr_g, p_postr_b, stream);

    // scratchpad read
    sutra_gemm(mu, p_sp_read_w, p_sp_read_b, qb, cM, cDIM, cDIM, 0, ACT_NONE, 1.f, stream);
    sutra_sp_read_kernel<<<cM / 8, 256, 0, stream>>>(qb, mem_cur, mem_ctx);
    sutra_add_memctx_kernel<<<MDgrid, 256, 0, stream>>>(mem_ctx, pi, msg);

    // precision-weighted write into mu
    sutra_layernorm(mu, mu_n, p_prew_g, p_prew_b, stream);
    sutra_gemm(msg, p_wr_lam_w, p_wr_lam_b, lam_obs, cM, cDIM, cDIM, 0, ACT_SOFTPLUS, 1.f, stream);
    sutra_gemm(msg, p_wr_val_w, p_wr_val_b, val,     cM, cDIM, cDIM, 0, ACT_NONE,     1.f, stream);
    sutra_mu_lam_update_kernel<<<MDgrid, 256, 0, stream>>>(mu_n, lam_obs, val, pi, mu, lam);
    sutra_layernorm(mu, mu, p_postw_g, p_postw_b, stream);
    sutra_add_so_kernel<<<MDgrid, 256, 0, stream>>>(so, mu);

    // scratchpad write (ping-pong: sp_write reads full old mem)
    sutra_sp_summ_kernel<<<cB, 256, 0, stream>>>(mu, pi, summ);
    sutra_sp_write_kernel<<<(cB * cNSLOT * cDIM + 255) / 256, 256, 0, stream>>>(
        mem_cur, summ, p_sp_gate_w, p_sp_gate_b, p_sp_val_w, p_sp_val_b, mem_nxt);
    float* t2 = mem_cur; mem_cur = mem_nxt; mem_nxt = t2;
  }

  // ---------------- final projection to vocab ----------------
  sutra_layernorm(mu, mu_n, p_ln_g, p_ln_b, stream);
  sutra_gemm(mu_n, p_emb, nullptr, (float*)d_out, cM, cVOCAB, cDIM,
             /*transB=*/1, ACT_NONE, cRSQRT_DIM, stream);
}
